// DotProductAttention_1022202217284
// MI455X (gfx1250) — compile-verified, matrix operand order checked by reading, not run on hardware
//
#include <hip/hip_runtime.h>
#include <math.h>

typedef __attribute__((ext_vector_type(16))) __bf16 v16bf;
typedef __attribute__((ext_vector_type(8)))  float  v8f;
typedef unsigned short ushort_t;

#define B_  16
#define N_  2048
#define D_  64
#define QT  16      // q rows per block
#define KT  32      // keys per iteration

// ---- float -> bf16 bits (native v_cvt_pk_bf16_f32 path) ----
__device__ __forceinline__ ushort_t f2bfbits(float f) {
    __bf16 h = (__bf16)f;
    union { __bf16 b; ushort_t s; } c; c.b = h;
    return c.s;
}

// ---- build a v16bf WMMA fragment from two contiguous 16B chunks ----
union FragU { uint4 u[2]; v16bf v; };
__device__ __forceinline__ v16bf load_frag16(const ushort_t* base, int off0, int off1) {
    FragU f;
    f.u[0] = *(const uint4*)(base + off0);
    f.u[1] = *(const uint4*)(base + off1);
    return f.v;
}

// K fragments for one 32-key iteration: [key subtile][K chunk]
struct KFrag { v16bf f[2][2]; };

__device__ __forceinline__ void load_kfrag(KFrag& kf, const ushort_t* Kbb,
                                           int k0, int lr, int lg) {
    #pragma unroll
    for (int t = 0; t < 2; ++t) {
        const ushort_t* krow = Kbb + (size_t)(k0 + t * 16 + lr) * D_;
        #pragma unroll
        for (int c = 0; c < 2; ++c)
            kf.f[t][c] = load_frag16(krow, c * 32 + lg * 8, c * 32 + lg * 8 + 16);
    }
}

// =====================================================================
// Pre-pass: K -> bf16 (same [b][n][d] layout), V -> bf16 transposed to
// Vt[b][d][n] so the second GEMM's B-fragments are contiguous loads.
// =====================================================================
__global__ __launch_bounds__(256)
void cvt_kv(const float* __restrict__ K, const float* __restrict__ V,
            ushort_t* __restrict__ Kb, ushort_t* __restrict__ Vt) {
    const size_t i = (size_t)blockIdx.x * blockDim.x + threadIdx.x; // 0 .. B*N*D-1
    Kb[i] = f2bfbits(K[i]);                       // both sides coalesced
    const size_t n = i & (N_ - 1);                // Vt index: ((b*D + d)*N + n)
    const size_t d = (i >> 11) & (D_ - 1);
    const size_t b = i >> 17;
    Vt[i] = f2bfbits(V[(b * N_ + n) * D_ + d]);   // write-coalesced transpose
}

// One pipelined iteration: compute from CUR's K frags, prefetch into NXT.
// Macro so register buffers keep compile-time names (no scratch demotion)
// and __shared__ arrays are referenced directly (ds_* lowering preserved).
#define ATTN_STEP(K0, CUR, NXT, DO_PF) do {                                     \
    /* V fragments for THIS step: consumed after 2 barriers (latency hidden) */ \
    v16bf vf[4];                                                                \
    _Pragma("unroll")                                                           \
    for (int t = 0; t < 4; ++t) {                                               \
        const ushort_t* vrow = Vtb + (size_t)(t * 16 + lr) * N_ + (K0);         \
        vf[t] = load_frag16(vrow, lg * 8, 16 + lg * 8);                         \
    }                                                                           \
    /* prefetch NEXT step's K fragments into the other named buffer */          \
    if (DO_PF) load_kfrag(NXT, Kbb, (K0) + KT, lr, lg);                         \
    /* scores: S = (Q/8) @ K^T, 16x32 per batch */                              \
    _Pragma("unroll")                                                           \
    for (int t = 0; t < 2; ++t) {                                               \
        v8f s = (v8f){0.f,0.f,0.f,0.f,0.f,0.f,0.f,0.f};                         \
        _Pragma("unroll")                                                       \
        for (int c = 0; c < 2; ++c)                                             \
            s = __builtin_amdgcn_wmma_f32_16x16x32_bf16(false, qf[c], false,    \
                    CUR.f[t][c], (short)0, s, false, false);                    \
        _Pragma("unroll")                                                       \
        for (int r = 0; r < 8; ++r)                                             \
            sS[((r + 8 * lg) * KT + t * 16 + lr) * B_ + w] = s[r];              \
    }                                                                           \
    __syncthreads();                                                            \
    /* softmax over the BATCH axis: 512 threads <-> 512 (q,k) positions */      \
    {                                                                           \
        const int tid = w * 32 + lane;                                          \
        const int q = tid >> 5, k = tid & 31;                                   \
        const float* p = &sS[(q * KT + k) * B_];                                \
        float vv[B_];                                                           \
        float m = -INFINITY;                                                    \
        _Pragma("unroll")                                                       \
        for (int b = 0; b < B_; ++b) { vv[b] = p[b]; m = fmaxf(m, vv[b]); }     \
        float sum = 0.f;                                                        \
        _Pragma("unroll")                                                       \
        for (int b = 0; b < B_; ++b) { vv[b] = __expf(vv[b] - m); sum += vv[b]; } \
        const float rs = 1.f / sum;                                             \
        _Pragma("unroll")                                                       \
        for (int b = 0; b < B_; ++b)                                            \
            sA[(b * QT + q) * KT + k] = f2bfbits(vv[b] * rs);                   \
    }                                                                           \
    __syncthreads();                                                            \
    /* out += attn @ V (16x32 @ 32x64 per batch) */                             \
    {                                                                           \
        const v16bf af = load_frag16(&sA[(w * QT + lr) * KT], lg*8, 16 + lg*8); \
        _Pragma("unroll")                                                       \
        for (int t = 0; t < 4; ++t)                                             \
            acc[t] = __builtin_amdgcn_wmma_f32_16x16x32_bf16(false, af, false,  \
                         vf[t], (short)0, acc[t], false, false);                \
    }                                                                           \
    /* no loop-end barrier: next sS scatter is ordered by the post-softmax  */   \
    /* barrier (all waves read sS before signaling it); sA writes by the next pre-softmax one.    */   \
} while (0)

// =====================================================================
// Main kernel: block = 16 waves, wave w owns batch b=w; block owns a
// 16-row q tile. Softmax couples only the 16 batch values per (q,k).
// =====================================================================
__global__ __launch_bounds__(512, 1)
void batch_softmax_attn(const float* __restrict__ Q,
                        const ushort_t* __restrict__ Kb,
                        const ushort_t* __restrict__ Vt,
                        float* __restrict__ O) {
    __shared__ float    sS[QT * KT * B_];   // raw scores [q][k][b]      (32 KB)
    __shared__ ushort_t sA[B_ * QT * KT];   // normalized bf16 [b][q][k] (16 KB)

    const int lane = threadIdx.x;        // 0..31
    const int w    = threadIdx.y;        // wave id == batch id (16 waves)
    const int lg   = lane >> 4;          // lane group (0/1) per WMMA layout
    const int lr   = lane & 15;
    const int q0   = blockIdx.x * QT;

    const float*    Qb  = Q  + ((size_t)w * N_ + q0) * D_;
    const ushort_t* Kbb = Kb + (size_t)w * N_ * D_;
    const ushort_t* Vtb = Vt + (size_t)w * D_ * N_;
    float*          Ob  = O  + ((size_t)w * N_ + q0) * D_;

    // ---- Q A-fragments (16x32 bf16 x 2 K-chunks), pre-scaled by 1/sqrt(64)
    v16bf qf[2];
    {
        const float* rowp = Qb + lr * D_;
        #pragma unroll
        for (int c = 0; c < 2; ++c) {
            const int base0 = c * 32 + lg * 8;
            const int base1 = base0 + 16;
            #pragma unroll
            for (int e = 0; e < 8; ++e) {
                qf[c][e]     = (__bf16)(rowp[base0 + e] * 0.125f);
                qf[c][e + 8] = (__bf16)(rowp[base1 + e] * 0.125f);
            }
        }
    }

    // output accumulators: 4 d-subtiles of 16 cols (f32 C/D layout)
    v8f acc[4];
    #pragma unroll
    for (int t = 0; t < 4; ++t) acc[t] = (v8f){0.f,0.f,0.f,0.f,0.f,0.f,0.f,0.f};

    // ---- two statically named K buffers: ping-pong with compile-time parity
    KFrag ka, kb;
    load_kfrag(ka, Kbb, 0, lr, lg);

    for (int k0 = 0; k0 < N_; k0 += 2 * KT) {
        ATTN_STEP(k0,      ka, kb, true);
        ATTN_STEP(k0 + KT, kb, ka, (k0 + 2 * KT) < N_);
    }

    // ---- store output (C/D layout -> row major), coalesced 64B per half-wave
    #pragma unroll
    for (int t = 0; t < 4; ++t) {
        #pragma unroll
        for (int r = 0; r < 8; ++r) {
            const int q = r + 8 * lg;
            const int d = t * 16 + lr;
            Ob[(size_t)q * D_ + d] = acc[t][r];
        }
    }
}

extern "C" void kernel_launch(void* const* d_in, const int* in_sizes, int n_in,
                              void* d_out, int out_size, void* d_ws, size_t ws_size,
                              hipStream_t stream) {
    const float* Q = (const float*)d_in[0];
    const float* K = (const float*)d_in[1];
    const float* V = (const float*)d_in[2];
    float* O = (float*)d_out;
    (void)in_sizes; (void)n_in; (void)out_size; (void)ws_size;

    ushort_t* Kb = (ushort_t*)d_ws;                       // 4 MiB
    ushort_t* Vt = Kb + (size_t)B_ * N_ * D_;             // 4 MiB

    const size_t total = (size_t)B_ * N_ * D_;            // 2M elements
    cvt_kv<<<dim3((unsigned)(total / 256)), dim3(256), 0, stream>>>(K, V, Kb, Vt);

    dim3 grid(N_ / QT);      // 128 q-tiles
    dim3 block(32, B_);      // 16 waves, wave w == batch b
    batch_softmax_attn<<<grid, block, 0, stream>>>(Q, Kb, Vt, O);
}
